// MaskedMultiSelfAttention_70119636075209
// MI455X (gfx1250) — matrix operand order repaired and checked by reading, compiled, not run on hardware
//
#include <hip/hip_runtime.h>
#include <hip/hip_bf16.h>

typedef __attribute__((ext_vector_type(16))) _Float16 v16h;
typedef __attribute__((ext_vector_type(8)))  _Float16 v8h;
typedef __attribute__((ext_vector_type(8)))  float    v8f;
typedef __attribute__((ext_vector_type(4)))  int      v4i;

// ---- CDNA5 async global->LDS path (guarded; falls back to sync staging) ----
#if defined(__has_builtin)
#if __has_builtin(__builtin_amdgcn_global_load_async_to_lds_b128) && \
    __has_builtin(__builtin_amdgcn_s_wait_asynccnt)
#define USE_ASYNC_LDS 1
#endif
#endif
#ifndef USE_ASYNC_LDS
#define USE_ASYNC_LDS 0
#endif

typedef __attribute__((address_space(1))) v4i gas_v4i;
typedef __attribute__((address_space(3))) v4i las_v4i;

static __device__ __forceinline__ void async_copy_b128(const void* gsrc, void* ldst) {
#if USE_ASYNC_LDS
    __builtin_amdgcn_global_load_async_to_lds_b128(
        (gas_v4i*)gsrc, (las_v4i*)ldst, 0, 0);
#else
    *(v8h*)ldst = *(const v8h*)gsrc;
#endif
}

static __device__ __forceinline__ void async_wait_all() {
#if USE_ASYNC_LDS
    __builtin_amdgcn_s_wait_asynccnt(0);
#endif
}

static __device__ __forceinline__ v16h combine16(v8h lo, v8h hi) {
    return __builtin_shufflevector(lo, hi, 0,1,2,3,4,5,6,7,8,9,10,11,12,13,14,15);
}

static __device__ __forceinline__ v8f wmma_f16(v16h a, v16h b, v8f c) {
    // D = A(16x32 f16) * B(32x16 f16) + C(16x16 f32)
    return __builtin_amdgcn_wmma_f32_16x16x32_f16(false, a, false, b, (short)0, c, false, false);
}

// ---------------------------------------------------------------------------
// f32 -> f16 conversion (4 elements / thread)
// ---------------------------------------------------------------------------
__global__ void cvt_f32_to_f16(const float* __restrict__ src,
                               _Float16* __restrict__ dst, int n) {
    int i = (blockIdx.x * blockDim.x + threadIdx.x) * 4;
    if (i + 4 <= n) {
        float4 v = *(const float4*)(src + i);
        dst[i + 0] = (_Float16)v.x;
        dst[i + 1] = (_Float16)v.y;
        dst[i + 2] = (_Float16)v.z;
        dst[i + 3] = (_Float16)v.w;
    } else {
        for (; i < n; ++i) dst[i] = (_Float16)src[i];
    }
}

// ---------------------------------------------------------------------------
// Tiled WMMA GEMM: C[M,N] = A[M,K](f16) * B[K,N](f16) + bias[N](f32)
// block = 256 threads (8 waves in 4x2), block tile = 128x128,
// wave tile = 32x64 (2x4 WMMA tiles), k-step = 32
// ---------------------------------------------------------------------------
template <bool OUT_F16>
__global__ __launch_bounds__(256)
void gemm_wmma(const _Float16* __restrict__ A,
               const _Float16* __restrict__ Bm,
               const float* __restrict__ bias,
               void* __restrict__ Cout,
               int M, int K, int N) {
    __shared__ _Float16 As[128][40];  // row-major A tile (+pad)
    __shared__ _Float16 Bs[128][40];  // TRANSPOSED B tile: Bs[n][k] (+pad)

    const int tid  = threadIdx.x;
    const int lane = tid & 31;
    const int wave = tid >> 5;
    const int wr   = wave >> 1;       // wave row (0..3) -> 32 rows each
    const int wc   = wave & 1;        // wave col (0..1) -> 64 cols each
    const int lh   = lane & 15;       // lane within 16-half
    const int hi   = lane >> 4;       // 0 or 1
    const int m0   = blockIdx.y * 128;
    const int n0   = blockIdx.x * 128;

    v8f acc[2][4] = {};

    #pragma unroll 2
    for (int kk = 0; kk < K; kk += 32) {
        // stage A: 128 rows x 32 halves; direct global->LDS async copies
        {
            int row = tid >> 1;
            int off = (tid & 1) * 16;
            const _Float16* src = A + (size_t)(m0 + row) * K + kk + off;
            async_copy_b128(src,     &As[row][off]);
            async_copy_b128(src + 8, &As[row][off + 8]);
        }
        // stage B transposed: rows kk..kk+31 of B, cols n0..n0+127 (scatter)
        {
            int r  = tid >> 3;            // k within tile (0..31)
            int cs = (tid & 7) * 16;      // n start within tile (0..112)
            const _Float16* src = Bm + (size_t)(kk + r) * N + n0 + cs;
            v8h b0 = *(const v8h*)(src);
            v8h b1 = *(const v8h*)(src + 8);
            #pragma unroll
            for (int i = 0; i < 8; ++i) Bs[cs + i][r]     = b0[i];
            #pragma unroll
            for (int i = 0; i < 8; ++i) Bs[cs + 8 + i][r] = b1[i];
        }
        if (kk + 32 < K) {
            __builtin_prefetch(A  + (size_t)(m0 + (tid >> 1)) * K + kk + 32, 0, 0);
            __builtin_prefetch(Bm + (size_t)(kk + 32 + (tid >> 3)) * N + n0, 0, 0);
        }
        async_wait_all();
        __syncthreads();

        // A fragments (ISA 16-bit A layout: lane<16 K in {c..c+7, c+16..c+23}, c = hi*8)
        v16h af[2], bf[4];
        #pragma unroll
        for (int i = 0; i < 2; ++i) {
            int row = wr * 32 + i * 16 + lh;
            int c   = hi * 8;
            af[i] = combine16(*(const v8h*)&As[row][c],
                              *(const v8h*)&As[row][c + 16]);
        }
        // B fragments (lane holds column n; K elements {kb..kb+7, kb+16..kb+23}, kb = hi*8)
        #pragma unroll
        for (int j = 0; j < 4; ++j) {
            int col = wc * 64 + j * 16 + lh;
            int kb  = hi * 8;
            bf[j] = combine16(*(const v8h*)&Bs[col][kb],
                              *(const v8h*)&Bs[col][kb + 16]);
        }
        #pragma unroll
        for (int i = 0; i < 2; ++i)
            #pragma unroll
            for (int j = 0; j < 4; ++j)
                acc[i][j] = wmma_f16(af[i], bf[j], acc[i][j]);
        __syncthreads();
    }

    // epilogue: bias + store (C layout: VGPR v -> row v + hi*8, col = lane&15)
    #pragma unroll
    for (int i = 0; i < 2; ++i) {
        #pragma unroll
        for (int j = 0; j < 4; ++j) {
            int col  = n0 + wc * 64 + j * 16 + lh;
            float bv = bias ? bias[col] : 0.0f;
            #pragma unroll
            for (int v = 0; v < 8; ++v) {
                int row   = m0 + wr * 32 + i * 16 + v + hi * 8;
                float val = acc[i][j][v] + bv;
                if (OUT_F16)
                    ((_Float16*)Cout)[(size_t)row * N + col] = (_Float16)val;
                else
                    ((float*)Cout)[(size_t)row * N + col] = val;
            }
        }
    }
}

// ---------------------------------------------------------------------------
// Flash attention (causal), f16 in / f16 out, f32 accumulate.
// grid = (T/64, NH, B); block = 128 (4 waves); wave owns 16 query rows.
// ---------------------------------------------------------------------------
__global__ __launch_bounds__(128)
void attn_kernel(const _Float16* __restrict__ qkv,   // [B*T][3C]
                 _Float16* __restrict__ outAttn) {   // [B*T][C]
    const int T = 2048, C = 1024, D = 64;
    const int h  = blockIdx.y;
    const int b  = blockIdx.z;
    const int rb = blockIdx.x;

    __shared__ _Float16 Ks[32][72];        // K tile: [key][d]
    __shared__ _Float16 Vt[64][40];        // V tile transposed: [d][key]
    __shared__ _Float16 Ps[4][16][40];     // per-wave P tile: [row][key]

    const int tid  = threadIdx.x;
    const int lane = tid & 31;
    const int wave = tid >> 5;
    const int lh   = lane & 15;
    const int hi   = lane >> 4;
    const int qr0  = rb * 64 + wave * 16;
    const size_t rs   = (size_t)3 * C;
    const size_t base = (size_t)b * T * rs;

    // Load Q fragments for this wave's 16 rows (two 32-wide d chunks)
    v16h aq[2];
    {
        const _Float16* qp = qkv + base + (size_t)(qr0 + lh) * rs + h * D;
        #pragma unroll
        for (int f = 0; f < 2; ++f) {
            int c = f * 32 + hi * 8;
            aq[f] = combine16(*(const v8h*)(qp + c), *(const v8h*)(qp + c + 16));
        }
    }

    float mrow[8], lrow[8];
    v8f acc[4] = {};
    #pragma unroll
    for (int v = 0; v < 8; ++v) { mrow[v] = -1e30f; lrow[v] = 0.0f; }

    const int nkt = rb * 2 + 2;            // causal: key tiles of 32 needed
    for (int jt = 0; jt < nkt; ++jt) {
        const int t0 = jt * 32;
        // K tile: async global->LDS; V tile: transpose via registers
        {
            int key = tid >> 2;
            int ds  = (tid & 3) * 16;
            const _Float16* kp = qkv + base + (size_t)(t0 + key) * rs + C + h * D + ds;
            async_copy_b128(kp,     &Ks[key][ds]);
            async_copy_b128(kp + 8, &Ks[key][ds + 8]);
            const _Float16* vp = qkv + base + (size_t)(t0 + key) * rs + 2 * C + h * D + ds;
            v8h v0 = *(const v8h*)vp;
            v8h v1 = *(const v8h*)(vp + 8);
            #pragma unroll
            for (int i = 0; i < 8; ++i) Vt[ds + i][key]     = v0[i];
            #pragma unroll
            for (int i = 0; i < 8; ++i) Vt[ds + 8 + i][key] = v1[i];
        }
        async_wait_all();
        __syncthreads();

        // S = Q * K^T  (two 16-col subtiles x two 32-wide d steps)
        v8f s[2] = {};
        #pragma unroll
        for (int tn = 0; tn < 2; ++tn) {
            #pragma unroll
            for (int ks = 0; ks < 2; ++ks) {
                int key = tn * 16 + lh;
                int kb  = ks * 32 + hi * 8;
                v16h bk = combine16(*(const v8h*)&Ks[key][kb],
                                    *(const v8h*)&Ks[key][kb + 16]);
                s[tn] = wmma_f16(aq[ks], bk, s[tn]);
            }
        }

        // online softmax update (rows live across 16 lanes; reduce with shfl_xor)
        const float scale = 0.125f;        // 1/sqrt(64)
        #pragma unroll
        for (int v = 0; v < 8; ++v) {
            int row = qr0 + v + hi * 8;
            int c0  = t0 + lh;
            float x0 = s[0][v] * scale + ((c0 <= row)      ? 0.0f : -1e30f);
            float x1 = s[1][v] * scale + ((c0 + 16 <= row) ? 0.0f : -1e30f);
            float tmax = fmaxf(x0, x1);
            #pragma unroll
            for (int msk = 1; msk < 16; msk <<= 1)
                tmax = fmaxf(tmax, __shfl_xor(tmax, msk, 16));
            float nm    = fmaxf(mrow[v], tmax);
            float alpha = __expf(mrow[v] - nm);
            float p0    = __expf(x0 - nm);
            float p1    = __expf(x1 - nm);
            float rsum  = p0 + p1;
            #pragma unroll
            for (int msk = 1; msk < 16; msk <<= 1)
                rsum += __shfl_xor(rsum, msk, 16);
            lrow[v] = lrow[v] * alpha + rsum;
            mrow[v] = nm;
            #pragma unroll
            for (int n = 0; n < 4; ++n) acc[n][v] *= alpha;
            // stash P (C-layout -> LDS) for re-read in A-layout
            Ps[wave][v + hi * 8][lh]      = (_Float16)p0;
            Ps[wave][v + hi * 8][16 + lh] = (_Float16)p1;
        }

        // P (16x32) as A fragment; O += P * V
        v16h ap;
        {
            int c = hi * 8;
            ap = combine16(*(const v8h*)&Ps[wave][lh][c],
                           *(const v8h*)&Ps[wave][lh][c + 16]);
        }
        #pragma unroll
        for (int n = 0; n < 4; ++n) {
            int d  = n * 16 + lh;
            int kb = hi * 8;
            v16h bv = combine16(*(const v8h*)&Vt[d][kb],
                                *(const v8h*)&Vt[d][kb + 16]);
            acc[n] = wmma_f16(ap, bv, acc[n]);
        }
        __syncthreads();
    }

    // normalize + store f16 [B*T][C], heads interleaved
    #pragma unroll
    for (int n = 0; n < 4; ++n) {
        int d = n * 16 + lh;
        #pragma unroll
        for (int v = 0; v < 8; ++v) {
            int row = qr0 + v + hi * 8;
            float val = acc[n][v] / lrow[v];
            outAttn[(size_t)(b * T + row) * C + h * D + d] = (_Float16)val;
        }
    }
}

// ---------------------------------------------------------------------------
extern "C" void kernel_launch(void* const* d_in, const int* in_sizes, int n_in,
                              void* d_out, int out_size, void* d_ws, size_t ws_size,
                              hipStream_t stream) {
    (void)in_sizes; (void)n_in; (void)out_size; (void)ws_size;
    const int B = 2, T = 2048, C = 1024;
    const int M = B * T;              // 4096
    const int N3 = 3 * C;             // 3072

    const float* x      = (const float*)d_in[0];
    const float* W_attn = (const float*)d_in[1];
    const float* b_attn = (const float*)d_in[2];
    const float* W_proj = (const float*)d_in[3];
    const float* b_proj = (const float*)d_in[4];
    float* out = (float*)d_out;

    // workspace partition (all f16)
    _Float16* x16    = (_Float16*)d_ws;
    _Float16* wa16   = x16    + (size_t)M * C;
    _Float16* wp16   = wa16   + (size_t)C * N3;
    _Float16* qkv16  = wp16   + (size_t)C * C;
    _Float16* attn16 = qkv16  + (size_t)M * N3;

    const int nx = M * C, nwa = C * N3, nwp = C * C;
    cvt_f32_to_f16<<<(nx / 4 + 255) / 256, 256, 0, stream>>>(x, x16, nx);
    cvt_f32_to_f16<<<(nwa / 4 + 255) / 256, 256, 0, stream>>>(W_attn, wa16, nwa);
    cvt_f32_to_f16<<<(nwp / 4 + 255) / 256, 256, 0, stream>>>(W_proj, wp16, nwp);

    // qkv = x @ W_attn + b_attn (f16 out)
    gemm_wmma<true><<<dim3(N3 / 128, M / 128), dim3(256), 0, stream>>>(
        x16, wa16, b_attn, (void*)qkv16, M, C, N3);
    // causal flash attention
    attn_kernel<<<dim3(T / 64, 16, B), dim3(128), 0, stream>>>(qkv16, attn16);
    // out = attn @ W_proj + b_proj (f32 out)
    gemm_wmma<false><<<dim3(C / 128, M / 128), dim3(256), 0, stream>>>(
        attn16, wp16, b_proj, (void*)out, M, C, C);
}
